// EmbDistillationLoss_44263932953090
// MI455X (gfx1250) — matrix operand order repaired and checked by reading, not compiled, not run on hardware
//
#include <hip/hip_runtime.h>
#include <hip/hip_bf16.h>
#include <stdint.h>

typedef __attribute__((ext_vector_type(16))) __bf16 v16bf;
typedef __attribute__((ext_vector_type(8)))  __bf16 v8bf;
typedef __attribute__((ext_vector_type(8)))  float  v8f;
typedef __attribute__((ext_vector_type(4)))  int    v4i;

#define BD 8
#define CD 512
#define TD 2048
#define KD 4096
#define ND (BD*TD)      // 16384 tokens
#define NTILES (KD/16)  // 256 column tiles

// Detect CDNA5 async global->LDS path (ASYNCcnt, CDNA5 ISA)
#if defined(__has_builtin)
#  if __has_builtin(__builtin_amdgcn_global_load_async_to_lds_b128) && \
      __has_builtin(__builtin_amdgcn_s_wait_asynccnt)
#    define USE_ASYNC_LDS 1
#  endif
#endif
#ifndef USE_ASYNC_LDS
#  define USE_ASYNC_LDS 0
#endif

// ---------------------------------------------------------------------------
// Kernel 1: gather/transpose student_emb (B,C,T)->(N,C), split f32 -> bf16
// hi/lo pair, per-row ||x||^2 and exact ||x - codebook[code]||^2.
// One wave32 per token row.
// ---------------------------------------------------------------------------
__global__ void __launch_bounds__(256)
prep_student(const float* __restrict__ emb,    // (B,C,T)
             const int*   __restrict__ codes,  // (B,T) flat
             const float* __restrict__ cb,     // (K,C)
             unsigned short* __restrict__ AhiU, unsigned short* __restrict__ AloU,
             float* __restrict__ x2, float* __restrict__ d2t) {
  __bf16* Ahi = (__bf16*)AhiU;
  __bf16* Alo = (__bf16*)AloU;
  int w    = (blockIdx.x * blockDim.x + threadIdx.x) >> 5;  // token row n
  int lane = threadIdx.x & 31;
  if (w >= ND) return;
  int b = w / TD, t = w % TD;
  int code = codes[w];
  const float* cbrow = cb + (size_t)code * CD;
  float sx2 = 0.f, sd2 = 0.f;
  for (int c = lane; c < CD; c += 32) {
    float x = emb[((size_t)b * CD + c) * TD + t];
    __bf16 hi = (__bf16)x;
    float  lo = x - (float)hi;
    Ahi[(size_t)w * CD + c] = hi;
    Alo[(size_t)w * CD + c] = (__bf16)lo;
    sx2 += x * x;
    float d = x - cbrow[c];
    sd2 += d * d;
  }
  for (int msk = 16; msk >= 1; msk >>= 1) {
    sx2 += __shfl_xor(sx2, msk, 32);
    sd2 += __shfl_xor(sd2, msk, 32);
  }
  if (lane == 0) { x2[w] = sx2; d2t[w] = sd2; }
}

// ---------------------------------------------------------------------------
// Kernel 2: split codebook -> bf16 hi/lo, per-code ||y||^2. One wave per code.
// ---------------------------------------------------------------------------
__global__ void __launch_bounds__(256)
prep_codebook(const float* __restrict__ cb,
              unsigned short* __restrict__ BhiU, unsigned short* __restrict__ BloU,
              float* __restrict__ y2) {
  __bf16* Bhi = (__bf16*)BhiU;
  __bf16* Blo = (__bf16*)BloU;
  int w    = (blockIdx.x * blockDim.x + threadIdx.x) >> 5;
  int lane = threadIdx.x & 31;
  if (w >= KD) return;
  const float* row = cb + (size_t)w * CD;
  float sy2 = 0.f;
  for (int c = lane; c < CD; c += 32) {
    float y = row[c];
    __bf16 hi = (__bf16)y;
    Bhi[(size_t)w * CD + c] = hi;
    Blo[(size_t)w * CD + c] = (__bf16)(y - (float)hi);
    sy2 += y * y;
  }
  for (int msk = 16; msk >= 1; msk >>= 1) sy2 += __shfl_xor(sy2, msk, 32);
  if (lane == 0) y2[w] = sy2;
}

// ---------------------------------------------------------------------------
// Kernel 3: fused distance GEMM (bf16x3 split, f32 accumulate) + online
// log-sum-exp + argmin over K.
// Workgroup: 4 waves, each owning 16 full rows (64 rows/WG). A operand is
// pinned in registers for the whole kernel; B tiles (16 cols x 512 k, hi+lo,
// 32KB) are double-buffered through LDS, staged asynchronously via
// GLOBAL_LOAD_ASYNC_TO_LDS_B128 (ASYNCcnt) when available.
// ---------------------------------------------------------------------------
__global__ void __launch_bounds__(128)
dist_softmax(const unsigned short* __restrict__ AhiU, const unsigned short* __restrict__ AloU,
             const unsigned short* __restrict__ BhiU, const unsigned short* __restrict__ BloU,
             const float* __restrict__ x2, const float* __restrict__ y2,
             float* __restrict__ lseOut, int* __restrict__ predOut) {
  const __bf16* Ahi = (const __bf16*)AhiU;
  const __bf16* Alo = (const __bf16*)AloU;
  const __bf16* Bhi = (const __bf16*)BhiU;
  const __bf16* Blo = (const __bf16*)BloU;

  // [buf][hi/lo][col][k] : 2*2*16*512*2B = 64KB
  __shared__ __align__(32) __bf16 sB[2][2][16][CD];

  const int tid  = threadIdx.x;
  const int lane = tid & 31;
  const int wave = tid >> 5;         // 0..3, each owns 16 rows
  const int rowBase = blockIdx.x * 64;
  const int hf  = lane >> 4;         // lane half
  const int l15 = lane & 15;

  // WMMA bf16 A-operand (16x32): lane holds row M=l15; K base 0 or 16.
  const int aRow = rowBase + wave * 16 + l15;
  const int akb  = hf ? 16 : 0;
  // WMMA bf16 B-operand (32x16): lane holds col N=l15; K base 0 or 16.
  const int bkb  = hf ? 16 : 0;

  const __bf16* aHiRow = Ahi + (size_t)aRow * CD;
  const __bf16* aLoRow = Alo + (size_t)aRow * CD;

  // Pin the full-depth A fragments in registers (invariant across all column tiles)
  v16bf ah[16], al[16];
  #pragma unroll
  for (int ki = 0; ki < 16; ++ki) {
    const int k = ki * 32;
    v8bf a0h = *(const v8bf*)(aHiRow + k + akb);
    v8bf a1h = *(const v8bf*)(aHiRow + k + akb + 8);
    v8bf a0l = *(const v8bf*)(aLoRow + k + akb);
    v8bf a1l = *(const v8bf*)(aLoRow + k + akb + 8);
    #pragma unroll
    for (int i = 0; i < 8; ++i) {
      ah[ki][i] = a0h[i]; ah[ki][8 + i] = a1h[i];
      al[ki][i] = a0l[i]; al[ki][8 + i] = a1l[i];
    }
  }

  float rx2[8];
  #pragma unroll
  for (int j = 0; j < 8; ++j)
    rx2[j] = x2[rowBase + wave * 16 + j + hf * 8];   // row of acc VGPR j

  float m[8], s[8], bv[8];
  int bi[8];
  #pragma unroll
  for (int j = 0; j < 8; ++j) { m[j] = -1e30f; s[j] = 0.f; bv[j] = 3.0e38f; bi[j] = 0x7fffffff; }

  char* ldsBase = (char*)&sB[0][0][0][0];   // 32KB per buffer

  // -------- stage tile 0 into buffer 0 --------
  {
    const char* srcH = (const char*)Bhi;          // colBase = 0
    const char* srcL = (const char*)Blo;
    char* dst = ldsBase;
#if USE_ASYNC_LDS
    for (int c = tid; c < 2048; c += 128) {
      int off = c * 16;
      const char* sp = (off < 16384) ? (srcH + off) : (srcL + off - 16384);
      __builtin_amdgcn_global_load_async_to_lds_b128(
          (v4i __attribute__((address_space(1)))*)(uintptr_t)sp,
          (v4i __attribute__((address_space(3)))*)(uintptr_t)(dst + off), 0, 0);
    }
    __builtin_amdgcn_s_wait_asynccnt(0);
#else
    for (int c = tid; c < 2048; c += 128) {
      int off = c * 16;
      const char* sp = (off < 16384) ? (srcH + off) : (srcL + off - 16384);
      *(float4*)(dst + off) = *(const float4*)sp;
    }
#endif
  }
  __syncthreads();

  for (int ct = 0; ct < NTILES; ++ct) {
    const int buf = ct & 1;
    const int colBase = ct * 16;
    const bool pf = (ct + 1 < NTILES);

#if USE_ASYNC_LDS
    // kick off next tile into the other buffer (ASYNCcnt-tracked)
    if (pf) {
      const char* srcH = (const char*)(Bhi + (size_t)(colBase + 16) * CD);
      const char* srcL = (const char*)(Blo + (size_t)(colBase + 16) * CD);
      char* dst = ldsBase + (buf ^ 1) * 32768;
      for (int c = tid; c < 2048; c += 128) {
        int off = c * 16;
        const char* sp = (off < 16384) ? (srcH + off) : (srcL + off - 16384);
        __builtin_amdgcn_global_load_async_to_lds_b128(
            (v4i __attribute__((address_space(1)))*)(uintptr_t)sp,
            (v4i __attribute__((address_space(3)))*)(uintptr_t)(dst + off), 0, 0);
      }
    }
#else
    float4 t[16];
    if (pf) {
      const char* srcH = (const char*)(Bhi + (size_t)(colBase + 16) * CD);
      const char* srcL = (const char*)(Blo + (size_t)(colBase + 16) * CD);
      #pragma unroll
      for (int c = 0; c < 16; ++c) {
        int off = (tid + c * 128) * 16;
        const char* sp = (off < 16384) ? (srcH + off) : (srcL + off - 16384);
        t[c] = *(const float4*)sp;
      }
    }
#endif

    // -------- compute 16(rows) x 16(cols) distance tile from LDS B --------
    const int bCol = colBase + l15;
    const float cy2 = y2[bCol];
    const __bf16* bh0 = &sB[buf][0][l15][bkb];
    const __bf16* bl0 = &sB[buf][1][l15][bkb];

    v8f acc = {};
    #pragma unroll
    for (int ki = 0; ki < 16; ++ki) {
      v16bf bh = *(const v16bf*)(bh0 + ki * 32);
      v16bf bl = *(const v16bf*)(bl0 + ki * 32);
      // x.y ~= hi.hi + hi.lo + lo.hi  (fp32-accurate split product)
      acc = __builtin_amdgcn_wmma_f32_16x16x32_bf16(false, ah[ki], false, bh, (short)0, acc, false, false);
      acc = __builtin_amdgcn_wmma_f32_16x16x32_bf16(false, ah[ki], false, bl, (short)0, acc, false, false);
      acc = __builtin_amdgcn_wmma_f32_16x16x32_bf16(false, al[ki], false, bh, (short)0, acc, false, false);
    }

    // acc VGPR j, lane l: row = rowBase+wave*16+j+8*hf, col = colBase+l15
    #pragma unroll
    for (int j = 0; j < 8; ++j) {
      float d2   = rx2[j] + cy2 - 2.0f * acc[j];
      float dist = sqrtf(fmaxf(d2, 0.f));
      // min(dist,index) across the 16 lanes of this half (also = -tile max logit)
      float v = dist; int ci = bCol;
      #pragma unroll
      for (int msk = 1; msk <= 8; msk <<= 1) {
        float ov = __shfl_xor(v,  msk, 32);
        int   oi = __shfl_xor(ci, msk, 32);
        if (ov < v || (ov == v && oi < ci)) { v = ov; ci = oi; }
      }
      if (v < bv[j] || (v == bv[j] && ci < bi[j])) { bv[j] = v; bi[j] = ci; }
      // online log-sum-exp of logits = -dist
      float nm = fmaxf(m[j], -v);
      float e  = __expf(-dist - nm);
      #pragma unroll
      for (int msk = 1; msk <= 8; msk <<= 1) e += __shfl_xor(e, msk, 32);
      s[j] = s[j] * __expf(m[j] - nm) + e;
      m[j] = nm;
    }

#if USE_ASYNC_LDS
    __builtin_amdgcn_s_wait_asynccnt(0);
    __syncthreads();
#else
    if (pf) {
      char* dst = ldsBase + (buf ^ 1) * 32768;
      #pragma unroll
      for (int c = 0; c < 16; ++c) {
        int off = (tid + c * 128) * 16;
        *(float4*)(dst + off) = t[c];
      }
    }
    __syncthreads();
#endif
  }

  // merge-free epilogue: each wave owns its 16 rows outright
  if (l15 == 0) {
    #pragma unroll
    for (int j = 0; j < 8; ++j) {
      int r = rowBase + wave * 16 + j + hf * 8;
      lseOut[r]  = m[j] + __logf(s[j]);
      predOut[r] = bi[j];
    }
  }
}

// ---------------------------------------------------------------------------
// Kernel 4: final scalar reductions -> 5 outputs.
// ---------------------------------------------------------------------------
__global__ void __launch_bounds__(1024)
finalize(const float* __restrict__ lse, const float* __restrict__ d2t,
         const int* __restrict__ pred, const int* __restrict__ codes,
         float* __restrict__ out) {
  __shared__ float sce[1024], smse[1024], sdist[1024], sacc[1024];
  int tid = threadIdx.x;
  float ce = 0.f, mse = 0.f, dist = 0.f, accc = 0.f;
  for (int n = tid; n < ND; n += 1024) {
    float dt = sqrtf(d2t[n]);
    ce   += lse[n] + dt;
    mse  += d2t[n];
    dist += dt;
    accc += (pred[n] == codes[n]) ? 1.f : 0.f;
  }
  sce[tid] = ce; smse[tid] = mse; sdist[tid] = dist; sacc[tid] = accc;
  __syncthreads();
  for (int st = 512; st >= 1; st >>= 1) {
    if (tid < st) {
      sce[tid] += sce[tid + st]; smse[tid] += smse[tid + st];
      sdist[tid] += sdist[tid + st]; sacc[tid] += sacc[tid + st];
    }
    __syncthreads();
  }
  if (tid == 0) {
    float embLoss = smse[0] / (float)((size_t)ND * CD);
    float ceLoss  = sce[0] / (float)ND;
    out[0] = embLoss + ceLoss;     // total_loss
    out[1] = embLoss;              // emb_to_codebook_loss
    out[2] = ceLoss;               // ce_loss
    out[3] = sacc[0] / (float)ND;  // token_accuracy
    out[4] = sdist[0] / (float)ND; // emb_to_target_dist
  }
}

extern "C" void kernel_launch(void* const* d_in, const int* in_sizes, int n_in,
                              void* d_out, int out_size, void* d_ws, size_t ws_size,
                              hipStream_t stream) {
  const float* emb   = (const float*)d_in[0];   // (B,C,T) f32
  const int*   codes = (const int*)d_in[1];     // (B,T)   int
  const float* cb    = (const float*)d_in[2];   // (K,C)   f32
  // d_in[3] (distance_matrix) is unused by the reference computation.

  char* ws = (char*)d_ws;
  unsigned short* Ahi = (unsigned short*)ws; ws += (size_t)ND * CD * 2;
  unsigned short* Alo = (unsigned short*)ws; ws += (size_t)ND * CD * 2;
  unsigned short* Bhi = (unsigned short*)ws; ws += (size_t)KD * CD * 2;
  unsigned short* Blo = (unsigned short*)ws; ws += (size_t)KD * CD * 2;
  float* x2  = (float*)ws; ws += (size_t)ND * 4;
  float* y2  = (float*)ws; ws += (size_t)KD * 4;
  float* d2t = (float*)ws; ws += (size_t)ND * 4;
  float* lse = (float*)ws; ws += (size_t)ND * 4;
  int*   pred = (int*)ws;

  prep_student <<<ND / 8, 256, 0, stream>>>(emb, codes, cb, Ahi, Alo, x2, d2t);
  prep_codebook<<<KD / 8, 256, 0, stream>>>(cb, Bhi, Blo, y2);
  dist_softmax <<<ND / 64, 128, 0, stream>>>(Ahi, Alo, Bhi, Blo, x2, y2, lse, pred);
  finalize     <<<1, 1024, 0, stream>>>(lse, d2t, pred, codes, (float*)d_out);
}
